// Transformer_72499047956750
// MI455X (gfx1250) — matrix operand order repaired and checked by reading, not compiled, not run on hardware
//
#include <hip/hip_runtime.h>
#include <hip/hip_bf16.h>
#include <stdint.h>

// ---------------- problem constants (fixed by the reference) ----------------
// L=6, B=32, S=512, D=512, H=8, DK=64, FF=2048.  All GEMM dims are multiples
// of 64, so the 64x64 tiling needs no bounds checks.
static constexpr int   Lc  = 6;
static constexpr int   Bc  = 32;
static constexpr int   Sc  = 512;
static constexpr int   Dc  = 512;
static constexpr int   Hc  = 8;
static constexpr int   DKc = 64;
static constexpr int   FFc = 2048;
static constexpr int   NTOK = Bc * Sc;            // 16384 rows of the residual stream
static constexpr float EPSc = 1e-5f;
static constexpr float GELU_C = 0.7978845608028654f;
static constexpr float INV_SQRT_DK = 0.125f;      // 1/sqrt(64)

// ---------------- WMMA types (CDNA5 / gfx1250, wave32) ----------------------
typedef __attribute__((ext_vector_type(16))) __bf16 v16bf;
typedef __attribute__((ext_vector_type(8)))  float  v8f;

union Frag {
  uint32_t w[8];
  uint4    q[2];
  v16bf    v;
};

__device__ __forceinline__ uint32_t f2bf1(float f) {
  uint32_t u = __float_as_uint(f);
  return (u + 0x7FFFu + ((u >> 16) & 1u)) >> 16;   // round-to-nearest-even
}
__device__ __forceinline__ uint32_t pack2bf(float a, float b) {
  return f2bf1(a) | (f2bf1(b) << 16);
}
__device__ __forceinline__ float gelu_f(float x) {
  float x3 = x * x * x;
  return 0.5f * x * (1.0f + tanhf(GELU_C * (x + 0.044715f * x3)));
}

// ---------------- GEMM epilogue modes ---------------------------------------
enum { EP_PLAIN = 0, EP_BIAS = 1, EP_BIAS_GELU = 2, EP_QKV_PERM = 3, EP_O_PERM = 4 };

// C[M,N] = A[M,K] * B (+epilogue).  A fp32 row-major (lda=K).
// BT=false: B fp32 row-major (K,N).  BT=true: B fp32 row-major (N,K) -> A@B^T.
// Batched via blockIdx.z with element strides sA/sB/sC.
// Tile: 64x64 per block (128 thr = 4 waves, each wave 32x32 = 2x2 WMMA frags).
template <int MODE, bool BT>
__global__ __launch_bounds__(128) void gemm_wmma_k(
    const float* __restrict__ A, const float* __restrict__ Bm,
    const float* __restrict__ bias, float* __restrict__ C,
    int M, int N, int K, long long sA, long long sB, long long sC)
{
  // [64 rows][16 k-pairs + 4 pad] dwords; 80B row stride keeps b128 alignment
  __shared__ __align__(16) uint32_t ldsA[64 * 20];
  __shared__ __align__(16) uint32_t ldsB[64 * 20];

  const int z = blockIdx.z;
  A  += (long long)z * sA;
  Bm += (long long)z * sB;
  if constexpr (MODE == EP_PLAIN || MODE == EP_BIAS || MODE == EP_BIAS_GELU)
    C += (long long)z * sC;

  const int m0   = blockIdx.y * 64;
  const int n0   = blockIdx.x * 64;
  const int t    = threadIdx.x;
  const int lane = t & 31;
  const int wave = t >> 5;
  const int wm   = wave >> 1, wn = wave & 1;
  const int lrow = lane & 15;
  const int hl   = lane >> 4;

  v8f acc[2][2] = {};

  for (int kt = 0; kt < K; kt += 32) {
    // ---- stage A tile: 64 rows x 32 k (as 16 bf16-pairs/row) ----
#pragma unroll
    for (int i = 0; i < 8; ++i) {
      int u = t + 128 * i;
      int r = u >> 4, p = u & 15;
      const float* s = A + (size_t)(m0 + r) * K + (kt + 2 * p);
      ldsA[r * 20 + p] = pack2bf(s[0], s[1]);
    }
    // ---- stage B tile transposed into [n][kpair] ----
    if constexpr (BT) {
#pragma unroll
      for (int i = 0; i < 8; ++i) {
        int u = t + 128 * i;
        int r = u >> 4, p = u & 15;
        const float* s = Bm + (size_t)(n0 + r) * K + (kt + 2 * p);
        ldsB[r * 20 + p] = pack2bf(s[0], s[1]);
      }
    } else {
#pragma unroll
      for (int i = 0; i < 8; ++i) {
        int u = t + 128 * i;
        int n = u & 63, p = u >> 6;
        const float* s0 = Bm + (size_t)(kt + 2 * p) * N + (n0 + n);
        ldsB[n * 20 + p] = pack2bf(s0[0], s0[N]);
      }
    }
    __syncthreads();

    // ---- build fragments per ISA 16-bit layouts ----
    // A: lane half hl selects K {0..7,16..23} vs {8..15,24..31}; VGPR v holds
    //    pair p = (v<4?0:8) + hl*4 + (v&3)  -> two aligned b128 LDS reads.
    // B ([n][k] in LDS): same K split as A; VGPR v holds
    //    pair p = (v<4?0:8) + hl*4 + (v&3)  -> two aligned b128 LDS reads.
    Frag fa[2], fb[2];
#pragma unroll
    for (int f = 0; f < 2; ++f) {
      const uint32_t* ap = &ldsA[(wm * 32 + f * 16 + lrow) * 20];
      fa[f].q[0] = *(const uint4*)&ap[hl * 4];
      fa[f].q[1] = *(const uint4*)&ap[8 + hl * 4];
      const uint32_t* bp = &ldsB[(wn * 32 + f * 16 + lrow) * 20];
      fb[f].q[0] = *(const uint4*)&bp[hl * 4];
      fb[f].q[1] = *(const uint4*)&bp[8 + hl * 4];
    }
#pragma unroll
    for (int fi = 0; fi < 2; ++fi)
#pragma unroll
      for (int fj = 0; fj < 2; ++fj)
        acc[fi][fj] = __builtin_amdgcn_wmma_f32_16x16x32_bf16(
            false, fa[fi].v, false, fb[fj].v, (short)0, acc[fi][fj], false, false);
    __syncthreads();
  }

  // ---- epilogue: C layout lane = N, VGPR g = M (g + 8*hl) ----
#pragma unroll
  for (int fi = 0; fi < 2; ++fi) {
#pragma unroll
    for (int fj = 0; fj < 2; ++fj) {
      const int mBase = m0 + wm * 32 + fi * 16 + hl * 8;
      const int n     = n0 + wn * 32 + fj * 16 + lrow;
#pragma unroll
      for (int g = 0; g < 8; ++g) {
        const int m = mBase + g;
        float v = acc[fi][fj][g];
        if constexpr (MODE == EP_BIAS || MODE == EP_BIAS_GELU) v += bias[n];
        if constexpr (MODE == EP_BIAS_GELU) v = gelu_f(v);
        if constexpr (MODE == EP_PLAIN || MODE == EP_BIAS || MODE == EP_BIAS_GELU) {
          C[(size_t)m * N + n] = v;
        } else if constexpr (MODE == EP_QKV_PERM) {
          // reference: col c = dk*H + h ; scatter to (B,H,S,DK) contiguous
          int b = m >> 9, i2 = m & (Sc - 1);
          int dk = n >> 3, hh = n & (Hc - 1);
          C[(((size_t)(b * Hc + hh) * Sc) + i2) * DKc + dk] = v;
        } else { // EP_O_PERM: batched over z = b*H + h ; merge heads into (B*S, D)
          int b = z >> 3, hh = z & (Hc - 1);
          C[((size_t)(b * Sc + m)) * Dc + hh * DKc + n] = v;
        }
      }
    }
  }
}

// ---------------- softmax over rows of length 512 (in-place, with scale) ----
__global__ __launch_bounds__(128) void softmax_k(float* __restrict__ s)
{
  __shared__ float red[128];
  float* row = s + (size_t)blockIdx.x * Sc;
  const int t = threadIdx.x;
  float v[4], mx = -3.4e38f;
#pragma unroll
  for (int i = 0; i < 4; ++i) {
    v[i] = row[t + 128 * i] * INV_SQRT_DK;
    mx = fmaxf(mx, v[i]);
  }
  red[t] = mx; __syncthreads();
  for (int o = 64; o > 0; o >>= 1) { if (t < o) red[t] = fmaxf(red[t], red[t + o]); __syncthreads(); }
  mx = red[0]; __syncthreads();
  float sum = 0.f;
#pragma unroll
  for (int i = 0; i < 4; ++i) { v[i] = expf(v[i] - mx); sum += v[i]; }
  red[t] = sum; __syncthreads();
  for (int o = 64; o > 0; o >>= 1) { if (t < o) red[t] += red[t + o]; __syncthreads(); }
  float inv = 1.0f / red[0];
#pragma unroll
  for (int i = 0; i < 4; ++i) row[t + 128 * i] = v[i] * inv;
}

// ---------------- out = LayerNorm(h + f) * gamma + beta (rows of 512) -------
__global__ __launch_bounds__(128) void add_ln_k(
    const float* __restrict__ h, const float* __restrict__ f,
    const float* __restrict__ g, const float* __restrict__ b,
    float* __restrict__ out)
{
  __shared__ float red[128];
  const size_t r = (size_t)blockIdx.x * Dc;
  const int t = threadIdx.x;
  float zv[4], s = 0.f;
#pragma unroll
  for (int i = 0; i < 4; ++i) { zv[i] = h[r + t + 128 * i] + f[r + t + 128 * i]; s += zv[i]; }
  red[t] = s; __syncthreads();
  for (int o = 64; o > 0; o >>= 1) { if (t < o) red[t] += red[t + o]; __syncthreads(); }
  const float mean = red[0] * (1.0f / Dc); __syncthreads();
  float vs = 0.f;
#pragma unroll
  for (int i = 0; i < 4; ++i) { float d = zv[i] - mean; vs += d * d; }
  red[t] = vs; __syncthreads();
  for (int o = 64; o > 0; o >>= 1) { if (t < o) red[t] += red[t + o]; __syncthreads(); }
  const float rstd = rsqrtf(red[0] * (1.0f / Dc) + EPSc);
#pragma unroll
  for (int i = 0; i < 4; ++i) {
    int c = t + 128 * i;
    out[r + c] = g[c] * (zv[i] - mean) * rstd + b[c];
  }
}

// ---------------- host orchestration ----------------------------------------
extern "C" void kernel_launch(void* const* d_in, const int* in_sizes, int n_in,
                              void* d_out, int out_size, void* d_ws, size_t ws_size,
                              hipStream_t stream)
{
  (void)in_sizes; (void)n_in; (void)out_size; (void)ws_size;
  const float* x   = (const float*)d_in[0];
  const float* WQ  = (const float*)d_in[1];
  const float* WK  = (const float*)d_in[2];
  const float* WV  = (const float*)d_in[3];
  const float* WO  = (const float*)d_in[4];
  const float* K1  = (const float*)d_in[5];
  const float* B1  = (const float*)d_in[6];
  const float* K2  = (const float*)d_in[7];
  const float* B2  = (const float*)d_in[8];
  const float* G1  = (const float*)d_in[9];
  const float* BE1 = (const float*)d_in[10];
  const float* G2  = (const float*)d_in[11];
  const float* BE2 = (const float*)d_in[12];

  // workspace layout (floats)
  const size_t nAct = (size_t)NTOK * Dc;            // 8,388,608
  const size_t nHead = (size_t)Bc * Hc * Sc * DKc;  // 8,388,608
  float* ws  = (float*)d_ws;
  float* h   = ws;                 // residual stream
  float* Qp  = h   + nAct;         // (B,H,S,DK)
  float* Kp  = Qp  + nHead;
  float* Vp  = Kp  + nHead;
  float* att = Vp  + nHead;        // (B*S, D) merged heads
  float* tmp = att + nAct;         // GEMM outputs feeding residual adds
  float* sc  = tmp + nAct;         // (B*H, S, S) scores; reused as FFN hidden

  hipMemcpyAsync(h, x, nAct * sizeof(float), hipMemcpyDeviceToDevice, stream);

  const dim3 blk(128);
  for (int l = 0; l < Lc; ++l) {
    const float* wq  = WQ  + (size_t)l * Dc * Dc;
    const float* wk  = WK  + (size_t)l * Dc * Dc;
    const float* wv  = WV  + (size_t)l * Dc * Dc;
    const float* wo  = WO  + (size_t)l * Dc * Dc;
    const float* k1  = K1  + (size_t)l * Dc * FFc;
    const float* b1  = B1  + (size_t)l * FFc;
    const float* k2  = K2  + (size_t)l * FFc * Dc;
    const float* b2  = B2  + (size_t)l * Dc;
    const float* g1  = G1  + (size_t)l * Dc;
    const float* be1 = BE1 + (size_t)l * Dc;
    const float* g2  = G2  + (size_t)l * Dc;
    const float* be2 = BE2 + (size_t)l * Dc;

    // Q/K/V projections with head-permuting scatter to (B,H,S,DK)
    gemm_wmma_k<EP_QKV_PERM, false><<<dim3(8, 256, 1), blk, 0, stream>>>(
        h, wq, nullptr, Qp, NTOK, Dc, Dc, 0, 0, 0);
    gemm_wmma_k<EP_QKV_PERM, false><<<dim3(8, 256, 1), blk, 0, stream>>>(
        h, wk, nullptr, Kp, NTOK, Dc, Dc, 0, 0, 0);
    gemm_wmma_k<EP_QKV_PERM, false><<<dim3(8, 256, 1), blk, 0, stream>>>(
        h, wv, nullptr, Vp, NTOK, Dc, Dc, 0, 0, 0);

    // scores = Q @ K^T, batched over B*H
    gemm_wmma_k<EP_PLAIN, true><<<dim3(8, 8, Bc * Hc), blk, 0, stream>>>(
        Qp, Kp, nullptr, sc, Sc, Sc, DKc,
        (long long)Sc * DKc, (long long)Sc * DKc, (long long)Sc * Sc);

    softmax_k<<<Bc * Hc * Sc, blk, 0, stream>>>(sc);

    // attn = P @ V, scattered back to (B*S, H*DK)
    gemm_wmma_k<EP_O_PERM, false><<<dim3(1, 8, Bc * Hc), blk, 0, stream>>>(
        sc, Vp, nullptr, att, Sc, DKc, Sc,
        (long long)Sc * Sc, (long long)Sc * DKc, 0);

    // output projection
    gemm_wmma_k<EP_PLAIN, false><<<dim3(8, 256, 1), blk, 0, stream>>>(
        att, wo, nullptr, tmp, NTOK, Dc, Dc, 0, 0, 0);

    // h = LN(h + attn_out)
    add_ln_k<<<NTOK, blk, 0, stream>>>(h, tmp, g1, be1, h);

    // FFN: gelu(h @ K1 + b1) @ K2 + b2
    gemm_wmma_k<EP_BIAS_GELU, false><<<dim3(32, 256, 1), blk, 0, stream>>>(
        h, k1, b1, sc, NTOK, FFc, Dc, 0, 0, 0);
    gemm_wmma_k<EP_BIAS, false><<<dim3(8, 256, 1), blk, 0, stream>>>(
        sc, k2, b2, tmp, NTOK, Dc, FFc, 0, 0, 0);

    // h = LN(h + ffn_out); last layer writes d_out directly
    float* dst = (l == Lc - 1) ? (float*)d_out : h;
    add_ln_k<<<NTOK, blk, 0, stream>>>(h, tmp, g2, be2, dst);
  }
}
